// DAttentionBaseline_83279415869731
// MI455X (gfx1250) — compile-verified, matrix operand order checked by reading, not run on hardware
//
#include <hip/hip_runtime.h>
#include <hip/hip_bf16.h>
#include <math.h>

// ---------------------------------------------------------------------------
// Deformable attention pipeline for MI455X (gfx1250, wave32, WMMA 16x16x32 f16)
// All key-dimension buffers padded 900 -> 1024 so every WMMA GEMM runs guard-
// free with N=1024; softmax masking (P=0 for n>=900) makes padding harmless.
// q,k stored transposed (pos-major) so WMMA fragments load as 2x b128 per lane.
// Softmax passes are fully vectorized (ds_load_b128 / ds_store_b128).
// ---------------------------------------------------------------------------

typedef _Float16 v16h __attribute__((ext_vector_type(16)));
typedef _Float16 v8h  __attribute__((ext_vector_type(8)));
typedef float    v8f  __attribute__((ext_vector_type(8)));

#define BQ    8
#define NCH   384
#define WW    32
#define HWPX  1024
#define GRP   6
#define GCH   64
#define HEADS 12
#define DH    32
#define HK    15
#define NSAMP 225
#define LKEY  900          // 4 * NS (valid keys)
#define LPAD  928          // key count padded to multiple of 32 (attention)
#define LSTR  1024         // key row stride in memory (padded, guard-free)
#define NQ    1024

__device__ __forceinline__ float gelu_exact(float x) {
    return 0.5f * x * (1.0f + erff(x * 0.70710678118654752440f));
}
__device__ __forceinline__ int clampi(int v, int lo, int hi) {
    return v < lo ? lo : (v > hi ? hi : v);
}

// ---------------------------------------------------------------------------
// Elementwise f32 -> f16 convert
// ---------------------------------------------------------------------------
__global__ void convert_f16_kernel(const float* __restrict__ src,
                                   _Float16* __restrict__ dst, long n) {
    long e = (long)blockIdx.x * blockDim.x + threadIdx.x;
    if (e < n) dst[e] = (_Float16)src[e];
}

// ---------------------------------------------------------------------------
// Pack the 4 feature maps into the grouped offset-branch input (48, 256, 1024) f16
// ---------------------------------------------------------------------------
__global__ void pack_offset_input_kernel(const float* __restrict__ xr1,
                                         const float* __restrict__ xr2,
                                         const float* __restrict__ xx1,
                                         const float* __restrict__ xx2,
                                         _Float16* __restrict__ out) {
    long e = (long)blockIdx.x * blockDim.x + threadIdx.x;
    if (e >= (long)48 * 256 * HWPX) return;
    int p   = (int)(e & (HWPX - 1));
    long r  = e >> 10;
    int ch  = (int)(r & 255);
    int bg  = (int)(r >> 8);
    int b = bg / GRP, g = bg % GRP;
    int i = ch >> 6, cc = ch & 63;
    const float* img = (i == 0) ? xr1 : (i == 1) ? xr2 : (i == 2) ? xx1 : xx2;
    out[e] = (_Float16)img[((long)b * NCH + g * GCH + cc) * HWPX + p];
}

// ---------------------------------------------------------------------------
// Batched WMMA GEMM: Y[z] = W (MxK) @ X[z] (KxN) + bias.  N == 1024 always
// (padded), so staging and epilogue are guard-free.  Epilogue modes:
//   1: f16 out (row-major M x N)      2: f32 out + exact GELU
//   3: f32 out + residual             4: f16 out TRANSPOSED (N x M)
// 256 threads = 8 waves; tile 64(M) x 128(N); K stepped by 32 via LDS.
// ---------------------------------------------------------------------------
#define LDA 40    // halves; 80B row, 16B aligned chunks
#define LDB 136   // halves; 272B row, 16B aligned chunks

template <int OUT_MODE>
__global__ __launch_bounds__(256)
void gemm_wmma_kernel(const _Float16* __restrict__ W,
                      const _Float16* __restrict__ X,
                      const float* __restrict__ bias,
                      const float* __restrict__ resid,
                      void* __restrict__ Y,
                      int M, int N, int K, long strideX, long strideY) {
    __shared__ __align__(16) _Float16 As[64 * LDA];
    __shared__ __align__(16) _Float16 Bs[32 * LDB];

    const int batch = blockIdx.z;
    const int m0 = blockIdx.y * 64;
    const int n0 = blockIdx.x * 128;
    const _Float16* Xb = X + (long)batch * strideX;

    const int tid  = threadIdx.x;
    const int lane = tid & 31;
    const int wave = tid >> 5;
    const int lrow = lane & 15;
    const int kb   = (lane >> 4) ? 8 : 0;   // K sub-block per lane half (ISA layout)
    const int mr0  = (lane >> 4) ? 8 : 0;

    v8f c[4] = {};

    for (int k0 = 0; k0 < K; k0 += 32) {
        // Stage A tile 64x32: one v8h (16B) per thread
        {
            int mm = tid >> 2, kseg = (tid & 3) * 8;
            *(v8h*)&As[mm * LDA + kseg] =
                *(const v8h*)&W[(long)(m0 + mm) * K + k0 + kseg];
        }
        // Stage B tile 32x128: two v8h per thread, coalesced along n
#pragma unroll
        for (int it = 0; it < 2; ++it) {
            int idx = tid + it * 256;          // 0..511
            int kk = idx >> 4, nseg = (idx & 15) * 8;
            *(v8h*)&Bs[kk * LDB + nseg] =
                *(const v8h*)&Xb[(long)(k0 + kk) * N + n0 + nseg];
        }
        __syncthreads();

        // B fragment (32x16 strip for this wave): pairs run along K rows
        v16h bfrag;
#pragma unroll
        for (int v = 0; v < 8; ++v) {
            int kk = (v < 4) ? (kb + 2 * v) : (kb + 16 + 2 * (v - 4));
            bfrag[2 * v]     = Bs[kk * LDB + wave * 16 + lrow];
            bfrag[2 * v + 1] = Bs[(kk + 1) * LDB + wave * 16 + lrow];
        }
#pragma unroll
        for (int mb = 0; mb < 4; ++mb) {
            // A fragment: per-lane K runs are contiguous -> 2x ds_load_b128
            v8h a0 = *(const v8h*)&As[(mb * 16 + lrow) * LDA + kb];
            v8h a1 = *(const v8h*)&As[(mb * 16 + lrow) * LDA + kb + 16];
            v16h afrag;
#pragma unroll
            for (int i = 0; i < 8; ++i) { afrag[i] = a0[i]; afrag[8 + i] = a1[i]; }
            c[mb] = __builtin_amdgcn_wmma_f32_16x16x32_f16(
                false, afrag, false, bfrag, (short)0, c[mb], false, false);
        }
        __syncthreads();
    }

    // Epilogue. C/D layout: lane holds (m = r + mr0, n = lane&15)
    const int n = n0 + wave * 16 + lrow;
    if (OUT_MODE == 4) {
        // transposed f16 out: Y[n * M + m]; 8 consecutive m -> one b128 store
        _Float16* Yt = (_Float16*)Y + (long)batch * strideY + (long)n * M;
#pragma unroll
        for (int mb = 0; mb < 4; ++mb) {
            int mbase = m0 + mb * 16 + mr0;
            v8h seg;
#pragma unroll
            for (int r = 0; r < 8; ++r)
                seg[r] = (_Float16)(c[mb][r] + bias[mbase + r]);
            *(v8h*)&Yt[mbase] = seg;
        }
    } else {
#pragma unroll
        for (int mb = 0; mb < 4; ++mb) {
#pragma unroll
            for (int r = 0; r < 8; ++r) {
                int m = m0 + mb * 16 + mr0 + r;
                float val = c[mb][r] + bias[m];
                long oidx = (long)batch * strideY + (long)m * N + n;
                if (OUT_MODE == 2) val = gelu_exact(val);
                if (OUT_MODE == 3) val += resid[oidx];
                if (OUT_MODE == 1) ((_Float16*)Y)[oidx] = (_Float16)val;
                else               ((float*)Y)[oidx]    = val;
            }
        }
    }
}

// ---------------------------------------------------------------------------
// Offset branch tail: depthwise 3x3 stride2 + GELU + 1x1(64->2) + tanh + ref
// ---------------------------------------------------------------------------
__global__ __launch_bounds__(256)
void offset_tail_kernel(const float* __restrict__ t1,   // (48, 64, 32, 32) f32
                        const float* __restrict__ w2,   // (64, 1, 3, 3)
                        const float* __restrict__ b2,   // (64,)
                        const float* __restrict__ w3,   // (2, 64)
                        float* __restrict__ pos) {      // (48, 225, 2): (gy, gx)
    int bg = blockIdx.x;
    int p  = threadIdx.x;
    if (p >= NSAMP) return;
    int py = p / HK, px = p % HK;
    const float* tb = t1 + (long)bg * GCH * HWPX;
    float acc0 = 0.0f, acc1 = 0.0f;
    for (int c = 0; c < GCH; ++c) {
        const float* tc = tb + (long)c * HWPX + (py * 2) * WW + px * 2;
        const float* wc = w2 + c * 9;
        float s = b2[c];
#pragma unroll
        for (int ky = 0; ky < 3; ++ky)
#pragma unroll
            for (int kx = 0; kx < 3; ++kx)
                s += tc[ky * WW + kx] * wc[ky * 3 + kx];
        s = gelu_exact(s);
        acc0 += w3[c] * s;
        acc1 += w3[GCH + c] * s;
    }
    const float oscale = 2.0f / 14.0f;   // ORF / (HK-1)
    float offy = tanhf(acc0) * oscale;
    float offx = tanhf(acc1) * oscale;
    float cy = (float)(py * 2 + 1) / 31.0f * 2.0f - 1.0f;
    float cx = (float)(px * 2 + 1) / 31.0f * 2.0f - 1.0f;
    pos[((long)bg * NSAMP + p) * 2 + 0] = fminf(fmaxf(offy + cy, -1.0f), 1.0f);
    pos[((long)bg * NSAMP + p) * 2 + 1] = fminf(fmaxf(offx + cx, -1.0f), 1.0f);
}

// ---------------------------------------------------------------------------
// Bilinear gather -> sampled f16 (8, 384, 1024): cols 900..1023 zero-filled
// ---------------------------------------------------------------------------
__global__ void bilinear_sample_kernel(const float* __restrict__ xr1,
                                       const float* __restrict__ xr2,
                                       const float* __restrict__ xx1,
                                       const float* __restrict__ xx2,
                                       const float* __restrict__ pos,
                                       _Float16* __restrict__ out) {
    long e = (long)blockIdx.x * blockDim.x + threadIdx.x;
    if (e >= (long)BQ * NCH * LSTR) return;
    int j   = (int)(e & (LSTR - 1));
    long r  = e >> 10;
    if (j >= LKEY) { out[e] = (_Float16)0.0f; return; }
    int c   = (int)(r % NCH);
    int b   = (int)(r / NCH);
    int i   = j / NSAMP, s = j % NSAMP;
    int g   = c >> 6;
    int bg  = b * GRP + g;
    float gy = pos[((long)bg * NSAMP + s) * 2 + 0];
    float gx = pos[((long)bg * NSAMP + s) * 2 + 1];
    float x = (gx + 1.0f) * 0.5f * 31.0f;
    float y = (gy + 1.0f) * 0.5f * 31.0f;
    float x0f = floorf(x), y0f = floorf(y);
    float wx = x - x0f, wy = y - y0f;
    int x0 = clampi((int)x0f, 0, 31), x1 = clampi(x0 + 1, 0, 31);
    int y0 = clampi((int)y0f, 0, 31), y1 = clampi(y0 + 1, 0, 31);
    const float* img = (i == 0) ? xr1 : (i == 1) ? xr2 : (i == 2) ? xx1 : xx2;
    const float* base = img + ((long)b * NCH + c) * HWPX;
    float v00 = base[y0 * WW + x0], v01 = base[y0 * WW + x1];
    float v10 = base[y1 * WW + x0], v11 = base[y1 * WW + x1];
    out[e] = (_Float16)(v00 * (1 - wy) * (1 - wx) + v01 * (1 - wy) * wx +
                        v10 * wy * (1 - wx) + v11 * wy * wx);
}

// ---------------------------------------------------------------------------
// Fused attention per (head-batch, 16-query tile).
//   qT: (B, 1024, 384) pos-major   kT: (B, 1024, 384) pos-major
//   v:  (B, 384, 1024) chan-major  out: (B, 384, 1024) chan-major f16
// All fragment loads are unconditional 2x b128 per lane (pad masked via P=0).
// Softmax max/exp passes use 8-wide LDS vectors with branchless tail masking.
// ---------------------------------------------------------------------------
__global__ __launch_bounds__(256)
void attention_kernel(const _Float16* __restrict__ qT,
                      const _Float16* __restrict__ kT,
                      const _Float16* __restrict__ vh,
                      _Float16* __restrict__ outh) {
    __shared__ __align__(16) _Float16 SP[16][LPAD];  // logits -> probabilities
    __shared__ float red[16][16];
    __shared__ float rowmax[16], rowinv[16];
    __shared__ float partial[8][16][16];

    const int hb = blockIdx.y;              // 0..95
    const int qt = blockIdx.x;              // 0..63
    const int b = hb / HEADS, head = hb % HEADS;
    const _Float16* qTb = qT + (long)b * NQ * NCH + head * DH;
    const _Float16* kTb = kT + (long)b * NQ * NCH + head * DH;
    const _Float16* vb  = vh + (long)(b * NCH + head * DH) * LSTR;

    const int tid  = threadIdx.x;
    const int lane = tid & 31;
    const int wave = tid >> 5;
    const int lrow = lane & 15;
    const int kbse = (lane >> 4) ? 8 : 0;
    const int mr0  = (lane >> 4) ? 8 : 0;

    // A fragment = q tile: A[m][kk] = qT[qm][kk]; contiguous -> 2x b128
    const int qm = qt * 16 + lrow;
    v8h q0 = *(const v8h*)&qTb[(long)qm * NCH + kbse];
    v8h q1 = *(const v8h*)&qTb[(long)qm * NCH + kbse + 16];
    v16h aq;
#pragma unroll
    for (int i = 0; i < 8; ++i) { aq[i] = q0[i]; aq[8 + i] = q1[i]; }

    // --- S = q^T k : one wmma per 16x16 tile (K = D = 32), tiles over waves ---
    const float scale = 0.17677669529663688f;  // 1/sqrt(32)
    for (int nt = wave; nt < LPAD / 16; nt += 8) {
        int n = nt * 16 + lrow;
        v8h k0 = *(const v8h*)&kTb[(long)n * NCH + kbse];
        v8h k1 = *(const v8h*)&kTb[(long)n * NCH + kbse + 16];
        v16h bk;
#pragma unroll
        for (int i = 0; i < 8; ++i) { bk[i] = k0[i]; bk[8 + i] = k1[i]; }
        v8f cs = {};
        cs = __builtin_amdgcn_wmma_f32_16x16x32_f16(
            false, aq, false, bk, (short)0, cs, false, false);
#pragma unroll
        for (int r = 0; r < 8; ++r)
            SP[mr0 + r][nt * 16 + lrow] = (_Float16)(cs[r] * scale);
    }
    __syncthreads();

    // --- softmax: thread (row = tid>>4, 8-wide col chunk = (tid&15)*8) ---
    const int srow = tid >> 4, scol = tid & 15;
    float lmax = -1e30f;
    for (int nb = scol * 8; nb < LPAD; nb += 128) {
        v8h chunk = *(const v8h*)&SP[srow][nb];        // ds_load_b128
#pragma unroll
        for (int i = 0; i < 8; ++i) {
            float val = (float)chunk[i];
            lmax = fmaxf(lmax, (nb + i < LKEY) ? val : -1e30f);
        }
    }
    red[srow][scol] = lmax;
    __syncthreads();
    if (scol == 0) {
        float m = red[srow][0];
        for (int i = 1; i < 16; ++i) m = fmaxf(m, red[srow][i]);
        rowmax[srow] = m;
    }
    __syncthreads();
    const float rm = rowmax[srow];
    float lsum = 0.0f;
    for (int nb = scol * 8; nb < LPAD; nb += 128) {
        v8h chunk = *(const v8h*)&SP[srow][nb];        // ds_load_b128
        v8h pc;
#pragma unroll
        for (int i = 0; i < 8; ++i) {
            float p = (nb + i < LKEY) ? __expf((float)chunk[i] - rm) : 0.0f;
            pc[i] = (_Float16)p;
            lsum += p;
        }
        *(v8h*)&SP[srow][nb] = pc;                     // ds_store_b128, in place
    }
    red[srow][scol] = lsum;
    __syncthreads();
    if (scol == 0) {
        float s = 0.0f;
        for (int i = 0; i < 16; ++i) s += red[srow][i];
        rowinv[srow] = 1.0f / s;
    }
    __syncthreads();

    // --- O = P @ v^T : wave -> (d-tile = wave&1, key chunks = wave>>1 step 4)
    const int dt = wave & 1;
    const _Float16* vrow = vb + (long)(dt * 16 + lrow) * LSTR;
    v8f co = {};
    for (int ch = (wave >> 1); ch < LPAD / 32; ch += 4) {
        int nb = ch * 32;
        v8h p0 = *(const v8h*)&SP[lrow][nb + kbse];        // ds_load_b128
        v8h p1 = *(const v8h*)&SP[lrow][nb + kbse + 16];
        v8h w0 = *(const v8h*)&vrow[nb + kbse];            // global b128
        v8h w1 = *(const v8h*)&vrow[nb + kbse + 16];
        v16h ap, bv;
#pragma unroll
        for (int i = 0; i < 8; ++i) {
            ap[i] = p0[i]; ap[8 + i] = p1[i];
            bv[i] = w0[i]; bv[8 + i] = w1[i];
        }
        co = __builtin_amdgcn_wmma_f32_16x16x32_f16(
            false, ap, false, bv, (short)0, co, false, false);
    }
#pragma unroll
    for (int r = 0; r < 8; ++r) partial[wave][mr0 + r][lrow] = co[r];
    __syncthreads();

    // Reduce 4 wave-partials per d-tile, normalize, store channel-major
    for (int e = tid; e < 512; e += 256) {
        int dtile = e >> 8;
        int m  = (e >> 4) & 15;
        int dd = e & 15;
        float s = partial[dtile][m][dd] + partial[dtile + 2][m][dd] +
                  partial[dtile + 4][m][dd] + partial[dtile + 6][m][dd];
        s *= rowinv[m];
        outh[(long)(b * NCH + head * DH + dtile * 16 + dd) * NQ + qt * 16 + m] =
            (_Float16)s;
    }
}

// ---------------------------------------------------------------------------
// Host launcher
// ---------------------------------------------------------------------------
extern "C" void kernel_launch(void* const* d_in, const int* in_sizes, int n_in,
                              void* d_out, int out_size, void* d_ws, size_t ws_size,
                              hipStream_t stream) {
    const float* query  = (const float*)d_in[0];
    const float* x_r1   = (const float*)d_in[1];
    const float* x_r2   = (const float*)d_in[2];
    const float* x_x1   = (const float*)d_in[3];
    const float* x_x2   = (const float*)d_in[4];
    const float* off_w1 = (const float*)d_in[5];
    const float* off_b1 = (const float*)d_in[6];
    const float* off_w2 = (const float*)d_in[7];
    const float* off_b2 = (const float*)d_in[8];
    const float* off_w3 = (const float*)d_in[9];
    const float* q_w = (const float*)d_in[10];
    const float* q_b = (const float*)d_in[11];
    const float* k_w = (const float*)d_in[12];
    const float* k_b = (const float*)d_in[13];
    const float* v_w = (const float*)d_in[14];
    const float* v_b = (const float*)d_in[15];
    const float* o_w = (const float*)d_in[16];
    const float* o_b = (const float*)d_in[17];

    size_t off = 0;
    auto carve = [&](size_t bytes) {
        size_t o = off;
        off += (bytes + 255) & ~(size_t)255;
        return (char*)d_ws + o;
    };
    _Float16* dataF16  = (_Float16*)carve((size_t)48 * 256 * HWPX * 2);
    float*    t1       = (float*)   carve((size_t)48 * GCH * HWPX * 4);
    float*    pos      = (float*)   carve((size_t)48 * NSAMP * 2 * 4);
    _Float16* w1h      = (_Float16*)carve((size_t)GCH * 256 * 2);
    _Float16* qwh      = (_Float16*)carve((size_t)NCH * NCH * 2);
    _Float16* kwh      = (_Float16*)carve((size_t)NCH * NCH * 2);
    _Float16* vwh      = (_Float16*)carve((size_t)NCH * NCH * 2);
    _Float16* owh      = (_Float16*)carve((size_t)NCH * NCH * 2);
    _Float16* queryh   = (_Float16*)carve((size_t)BQ * NCH * NQ * 2);
    _Float16* sampledh = (_Float16*)carve((size_t)BQ * NCH * LSTR * 2);
    _Float16* qTh      = (_Float16*)carve((size_t)BQ * NQ * NCH * 2);
    _Float16* kTh      = (_Float16*)carve((size_t)BQ * LSTR * NCH * 2);
    _Float16* vhh      = (_Float16*)carve((size_t)BQ * NCH * LSTR * 2);
    _Float16* attnh    = (_Float16*)carve((size_t)BQ * NCH * NQ * 2);

    auto cvt = [&](const float* s, _Float16* d, long n) {
        convert_f16_kernel<<<dim3((unsigned)((n + 255) / 256)), 256, 0, stream>>>(s, d, n);
    };
    cvt(off_w1, w1h, (long)GCH * 256);
    cvt(q_w, qwh, (long)NCH * NCH);
    cvt(k_w, kwh, (long)NCH * NCH);
    cvt(v_w, vwh, (long)NCH * NCH);
    cvt(o_w, owh, (long)NCH * NCH);
    cvt(query, queryh, (long)BQ * NCH * NQ);

    {   // pack offset-branch input
        long n = (long)48 * 256 * HWPX;
        pack_offset_input_kernel<<<dim3((unsigned)((n + 255) / 256)), 256, 0, stream>>>(
            x_r1, x_r2, x_x1, x_x2, dataF16);
    }

    // Offset conv1: 48 x [64x1024 = 64x256 @ 256x1024] + GELU -> t1 (f32)
    gemm_wmma_kernel<2><<<dim3(8, 1, 48), 256, 0, stream>>>(
        w1h, dataF16, off_b1, nullptr, t1,
        64, HWPX, 256, (long)256 * HWPX, (long)GCH * HWPX);

    // Depthwise + GELU + 1x1 + tanh + reference points -> sampling grid
    offset_tail_kernel<<<dim3(48), 256, 0, stream>>>(t1, off_w2, off_b2, off_w3, pos);

    {   // Bilinear gather -> sampled f16 (padded to 1024, zero-filled)
        long n = (long)BQ * NCH * LSTR;
        bilinear_sample_kernel<<<dim3((unsigned)((n + 255) / 256)), 256, 0, stream>>>(
            x_r1, x_r2, x_x1, x_x2, pos, sampledh);
    }

    // q projection -> transposed (B, 1024, 384) f16
    gemm_wmma_kernel<4><<<dim3(8, 6, 8), 256, 0, stream>>>(
        qwh, queryh, q_b, nullptr, qTh,
        NCH, NQ, NCH, (long)NCH * NQ, (long)NQ * NCH);
    // k projection -> transposed (B, 1024, 384) f16 (pad cols masked via P=0)
    gemm_wmma_kernel<4><<<dim3(8, 6, 8), 256, 0, stream>>>(
        kwh, sampledh, k_b, nullptr, kTh,
        NCH, LSTR, NCH, (long)NCH * LSTR, (long)LSTR * NCH);
    // v projection -> (B, 384, 1024) f16
    gemm_wmma_kernel<1><<<dim3(8, 6, 8), 256, 0, stream>>>(
        vwh, sampledh, v_b, nullptr, vhh,
        NCH, LSTR, NCH, (long)NCH * LSTR, (long)NCH * LSTR);

    // Fused attention: 96 head-batches x 64 query tiles
    attention_kernel<<<dim3(64, 96), 256, 0, stream>>>(qTh, kTh, vhh, attnh);

    // Output projection + bias + residual -> d_out (f32)
    gemm_wmma_kernel<3><<<dim3(8, 6, 8), 256, 0, stream>>>(
        owh, attnh, o_b, query, d_out,
        NCH, NQ, NCH, (long)NCH * NQ, (long)NCH * NQ);
}
